// WindowMultiHeadAttention_64175401337030
// MI455X (gfx1250) — matrix operand order repaired and checked by reading, not compiled, not run on hardware
//
#include <hip/hip_runtime.h>
#include <hip/hip_bf16.h>

typedef __attribute__((ext_vector_type(16))) _Float16 v16h;
typedef __attribute__((ext_vector_type(8)))  _Float16 v8h;
typedef __attribute__((ext_vector_type(4)))  _Float16 v4h;
typedef __attribute__((ext_vector_type(8)))  float    v8f;

#define WMMA_F16(a, b, c) \
  __builtin_amdgcn_wmma_f32_16x16x32_f16(false, (a), false, (b), (short)0, (c), false, false)

// ---- CDNA5 async global->LDS copy (ASYNCcnt path), 16B per lane --------------
__device__ __forceinline__ void async_ld_b128(const void* gptr, void* lptr) {
  asm volatile("global_load_async_to_lds_b128 %0, %1, off"
               :: "v"((unsigned int)(unsigned long long)(uintptr_t)lptr),
                  "v"((unsigned long long)(uintptr_t)gptr)
               : "memory");
}
__device__ __forceinline__ void wait_async0() {
#if __has_builtin(__builtin_amdgcn_s_wait_asynccnt)
  __builtin_amdgcn_s_wait_asynccnt(0);
#else
  asm volatile("s_wait_asynccnt 0" ::: "memory");
#endif
}

// ---------------------------------------------------------------- cast f32->f16
__global__ void __launch_bounds__(256) cast_f32_to_f16_v4(
    const float* __restrict__ src, _Float16* __restrict__ dst, int n4) {
  int i = blockIdx.x * blockDim.x + threadIdx.x;
  int stride = gridDim.x * blockDim.x;
  for (; i < n4; i += stride) {
    float4 f = ((const float4*)src)[i];
    v4h h;
    h[0] = (_Float16)f.x; h[1] = (_Float16)f.y;
    h[2] = (_Float16)f.z; h[3] = (_Float16)f.w;
    ((v4h*)dst)[i] = h;
  }
}

// ---------------------------------------------------------------- tiled WMMA GEMM
// C[m][n] = sum_k A[m][k] * W[n][k] (+ bias[n]);  K = 384, tile 64x64, 4 waves.
// Double-buffered LDS, async global->LDS staging.
#define GLDA 40   // LDS row stride (halves): 80B rows, 16B aligned chunks

__global__ void __launch_bounds__(128) gemm_qkv_f16(
    const _Float16* __restrict__ A,    // M x 384
    const _Float16* __restrict__ Wt,   // 1152 x 384 (row-major, = qkv_w)
    const float* __restrict__ bias,    // 1152
    _Float16* __restrict__ Cout,       // M x 1152 (f16)
    int Ntiles) {
  __shared__ __align__(16) _Float16 sA[2][64 * GLDA];
  __shared__ __align__(16) _Float16 sB[2][64 * GLDA];
  const int K = 384;
  int bid = blockIdx.x;
  int nt = bid % Ntiles, mt = bid / Ntiles;
  int m0 = mt * 64, n0 = nt * 64;
  int t = threadIdx.x, wave = t >> 5, lane = t & 31;
  int cn = lane & 15, hi = lane >> 4;

  // per-thread staging chunks: 2 x 8 halves for A and for B
  int r0 = t >> 2,         q0 = (t & 3) * 8;
  int r1 = (t + 128) >> 2, q1 = ((t + 128) & 3) * 8;
  const _Float16* gA0 = &A[(m0 + r0) * K + q0];
  const _Float16* gA1 = &A[(m0 + r1) * K + q1];
  const _Float16* gB0 = &Wt[(n0 + r0) * K + q0];
  const _Float16* gB1 = &Wt[(n0 + r1) * K + q1];

  v8f acc[4] = {};

  // prologue: fill buffer 0
  async_ld_b128(gA0, &sA[0][r0 * GLDA + q0]);
  async_ld_b128(gA1, &sA[0][r1 * GLDA + q1]);
  async_ld_b128(gB0, &sB[0][r0 * GLDA + q0]);
  async_ld_b128(gB1, &sB[0][r1 * GLDA + q1]);

  int cur = 0;
  for (int kb = 0; kb < K; kb += 32) {
    wait_async0();
    __syncthreads();                    // cur filled everywhere; cur^1 free
    if (kb + 32 < K) {
      int nb = cur ^ 1, kn = kb + 32;
      async_ld_b128(gA0 + kn, &sA[nb][r0 * GLDA + q0]);
      async_ld_b128(gA1 + kn, &sA[nb][r1 * GLDA + q1]);
      async_ld_b128(gB0 + kn, &sB[nb][r0 * GLDA + q0]);
      async_ld_b128(gB1 + kn, &sB[nb][r1 * GLDA + q1]);
    }
    // B fragment (32x16, [N][K] layout): lane n=cn, K = (hi?8:0)+0..7, (hi?8:0)+16..23
    v16h bfrag;
    {
      const _Float16* bp = &sB[cur][(wave * 16 + cn) * GLDA + (hi ? 8 : 0)];
      ((v8h*)&bfrag)[0] = *(const v8h*)bp;
      ((v8h*)&bfrag)[1] = *(const v8h*)(bp + 16);
    }
    int akb = hi ? 8 : 0;
    for (int i = 0; i < 4; ++i) {
      v16h afrag;
      const _Float16* ap = &sA[cur][(i * 16 + cn) * GLDA + akb];
      ((v8h*)&afrag)[0] = *(const v8h*)ap;        // K = akb .. akb+7
      ((v8h*)&afrag)[1] = *(const v8h*)(ap + 16); // K = akb+16 .. akb+23
      acc[i] = WMMA_F16(afrag, bfrag, acc[i]);
    }
    cur ^= 1;
  }
  int col = n0 + wave * 16 + cn;
  float bv = bias[col];
  int rbase = m0 + hi * 8;
  for (int i = 0; i < 4; ++i)
    for (int v = 0; v < 8; ++v)
      Cout[(rbase + i * 16 + v) * 1152 + col] = (_Float16)(acc[i][v] + bv);
}

__global__ void __launch_bounds__(128) gemm_proj_f32(
    const _Float16* __restrict__ A,    // M x 384
    const _Float16* __restrict__ Wt,   // 384 x 384 (= proj_w)
    const float* __restrict__ bias,    // 384
    float* __restrict__ Cout,          // M x 384 (f32, = d_out)
    int Ntiles) {
  __shared__ __align__(16) _Float16 sA[2][64 * GLDA];
  __shared__ __align__(16) _Float16 sB[2][64 * GLDA];
  const int K = 384;
  int bid = blockIdx.x;
  int nt = bid % Ntiles, mt = bid / Ntiles;
  int m0 = mt * 64, n0 = nt * 64;
  int t = threadIdx.x, wave = t >> 5, lane = t & 31;
  int cn = lane & 15, hi = lane >> 4;

  int r0 = t >> 2,         q0 = (t & 3) * 8;
  int r1 = (t + 128) >> 2, q1 = ((t + 128) & 3) * 8;
  const _Float16* gA0 = &A[(m0 + r0) * K + q0];
  const _Float16* gA1 = &A[(m0 + r1) * K + q1];
  const _Float16* gB0 = &Wt[(n0 + r0) * K + q0];
  const _Float16* gB1 = &Wt[(n0 + r1) * K + q1];

  v8f acc[4] = {};

  async_ld_b128(gA0, &sA[0][r0 * GLDA + q0]);
  async_ld_b128(gA1, &sA[0][r1 * GLDA + q1]);
  async_ld_b128(gB0, &sB[0][r0 * GLDA + q0]);
  async_ld_b128(gB1, &sB[0][r1 * GLDA + q1]);

  int cur = 0;
  for (int kb = 0; kb < K; kb += 32) {
    wait_async0();
    __syncthreads();
    if (kb + 32 < K) {
      int nb = cur ^ 1, kn = kb + 32;
      async_ld_b128(gA0 + kn, &sA[nb][r0 * GLDA + q0]);
      async_ld_b128(gA1 + kn, &sA[nb][r1 * GLDA + q1]);
      async_ld_b128(gB0 + kn, &sB[nb][r0 * GLDA + q0]);
      async_ld_b128(gB1 + kn, &sB[nb][r1 * GLDA + q1]);
    }
    v16h bfrag;
    {
      const _Float16* bp = &sB[cur][(wave * 16 + cn) * GLDA + (hi ? 8 : 0)];
      ((v8h*)&bfrag)[0] = *(const v8h*)bp;
      ((v8h*)&bfrag)[1] = *(const v8h*)(bp + 16);
    }
    int akb = hi ? 8 : 0;
    for (int i = 0; i < 4; ++i) {
      v16h afrag;
      const _Float16* ap = &sA[cur][(i * 16 + cn) * GLDA + akb];
      ((v8h*)&afrag)[0] = *(const v8h*)ap;
      ((v8h*)&afrag)[1] = *(const v8h*)(ap + 16);
      acc[i] = WMMA_F16(afrag, bfrag, acc[i]);
    }
    cur ^= 1;
  }
  int col = n0 + wave * 16 + cn;
  float bv = bias[col];
  int rbase = m0 + hi * 8;
  for (int i = 0; i < 4; ++i)
    for (int v = 0; v < 8; ++v)
      Cout[(rbase + i * 16 + v) * 384 + col] = acc[i][v] + bv;
}

// ---------------------------------------------------------------- fused window attention
// One block = one (window, head). 49 tokens padded to 64. 4 waves.
__global__ void __launch_bounds__(128) win_attn(
    const _Float16* __restrict__ qkv,        // [100352][1152]: [q|k|v] x head x dim
    const float* __restrict__ bias_table,    // [169][12]
    _Float16* __restrict__ outh) {           // [100352][384], de-windowed
  constexpr int LDQ = 40;   // Q/K tiles: 64 x 32 (+8 pad)
  constexpr int LDV = 72;   // V transposed: 32(d) x 64(n) (+8 pad)
  constexpr int LDP = 72;   // P: 64 x 64 (+8 pad)
  __shared__ __align__(16) _Float16 sQ[64 * LDQ];
  __shared__ __align__(16) _Float16 sK[64 * LDQ];
  __shared__ __align__(16) _Float16 sV[32 * LDV];
  __shared__ __align__(16) _Float16 sP[64 * LDP];
  __shared__ int   rowmap[64];
  __shared__ float sbias[169];

  int bid = blockIdx.x;
  int h   = bid % 12;
  int win = bid / 12;
  int b = win >> 6, wi = win & 63;
  int wy = wi >> 3, wx = wi & 7;

  int t = threadIdx.x;
  if (t < 64) {
    int r = -1;
    if (t < 49) {
      int ty = t / 7, tx = t - ty * 7;
      r = b * 3136 + (wy * 7 + ty) * 56 + (wx * 7 + tx);
    }
    rowmap[t] = r;
  }
  for (int i = t; i < 169; i += 128) sbias[i] = bias_table[i * 12 + h];
  __syncthreads();

  // ---- stage Q, K (row-major) and V (transposed [d][n]) into LDS
  {
    int n = t & 63, part = t >> 6;   // 16 halves per (n,part)
    int r = rowmap[n];
    uint4 z = {0, 0, 0, 0};
    uint4 q0 = z, q1 = z, k0 = z, k1 = z, v0 = z, v1 = z;
    if (r >= 0) {
      int base = r * 1152 + h * 32 + part * 16;
      const uint4* qp = (const uint4*)&qkv[base];
      const uint4* kp = (const uint4*)&qkv[base + 384];
      const uint4* vp = (const uint4*)&qkv[base + 768];
      q0 = qp[0]; q1 = qp[1];
      k0 = kp[0]; k1 = kp[1];
      v0 = vp[0]; v1 = vp[1];
    }
    *(uint4*)&sQ[n * LDQ + part * 16]     = q0;
    *(uint4*)&sQ[n * LDQ + part * 16 + 8] = q1;
    *(uint4*)&sK[n * LDQ + part * 16]     = k0;
    *(uint4*)&sK[n * LDQ + part * 16 + 8] = k1;
    _Float16 vv[16];
    *(uint4*)&vv[0] = v0; *(uint4*)&vv[8] = v1;
    for (int j = 0; j < 16; ++j)
      sV[(part * 16 + j) * LDV + n] = vv[j];   // transpose: [d][n]
  }
  __syncthreads();

  int wave = t >> 5, lane = t & 31;
  int cn = lane & 15, hi = lane >> 4;
  int akb = hi ? 8 : 0;

  // ---- S = Q * K^T  (wave handles M rows [wave*16, wave*16+16), all 4 N tiles)
  v16h aQ;
  {
    const _Float16* ap = &sQ[(wave * 16 + cn) * LDQ + akb];
    ((v8h*)&aQ)[0] = *(const v8h*)ap;
    ((v8h*)&aQ)[1] = *(const v8h*)(ap + 16);
  }
  v8f sAcc[4];
  for (int j = 0; j < 4; ++j) {
    v16h bK;
    const _Float16* bp = &sK[(j * 16 + cn) * LDQ + (hi ? 8 : 0)];
    ((v8h*)&bK)[0] = *(const v8h*)bp;
    ((v8h*)&bK)[1] = *(const v8h*)(bp + 16);
    v8f zero = {};
    sAcc[j] = WMMA_F16(aQ, bK, zero);
  }

  // ---- scale + relative-position bias + masked softmax (f32)
  const float scale = 0.17677669529663687f;  // 1/sqrt(32)
  int mbase = wave * 16 + hi * 8;            // C-layout: row = mbase+v, col = j*16+cn
  float p[4][8];
  for (int v = 0; v < 8; ++v) {
    int m = mbase + v;
    bool mvalid = (m < 49);
    int qy = mvalid ? m / 7 : 0;
    int qx = mvalid ? m - qy * 7 : 0;
    float val[4];
    for (int j = 0; j < 4; ++j) {
      int col = j * 16 + cn;
      float s = sAcc[j][v] * scale;
      if (col < 49) {
        if (mvalid) {
          int ky = col / 7, kx = col - ky * 7;
          s += sbias[(qy - ky + 6) * 13 + (qx - kx + 6)];
        }
      } else {
        s = -3.0e38f;   // mask padded keys
      }
      val[j] = s;
    }
    float rmax = fmaxf(fmaxf(val[0], val[1]), fmaxf(val[2], val[3]));
    for (int d = 1; d < 16; d <<= 1) rmax = fmaxf(rmax, __shfl_xor(rmax, d, 16));
    float rsum = 0.0f;
    for (int j = 0; j < 4; ++j) { val[j] = __expf(val[j] - rmax); rsum += val[j]; }
    for (int d = 1; d < 16; d <<= 1) rsum += __shfl_xor(rsum, d, 16);
    float inv = 1.0f / rsum;
    for (int j = 0; j < 4; ++j) p[j][v] = val[j] * inv;
  }
  for (int j = 0; j < 4; ++j)
    for (int v = 0; v < 8; ++v)
      sP[(mbase + v) * LDP + j * 16 + cn] = (_Float16)p[j][v];
  __syncthreads();

  // ---- O = P * V   (M=16 rows per wave, N=32, K=64 in two 32-steps)
  v8f oAcc[2] = {};
  for (int kb = 0; kb < 64; kb += 32) {
    v16h aP;
    const _Float16* ap = &sP[(wave * 16 + cn) * LDP + kb + akb];
    ((v8h*)&aP)[0] = *(const v8h*)ap;
    ((v8h*)&aP)[1] = *(const v8h*)(ap + 16);
    for (int j = 0; j < 2; ++j) {
      v16h bV;
      const _Float16* bp = &sV[(j * 16 + cn) * LDV + kb + (hi ? 8 : 0)];
      ((v8h*)&bV)[0] = *(const v8h*)bp;
      ((v8h*)&bV)[1] = *(const v8h*)(bp + 16);
      oAcc[j] = WMMA_F16(aP, bV, oAcc[j]);
    }
  }
  for (int j = 0; j < 2; ++j)
    for (int v = 0; v < 8; ++v) {
      int m = mbase + v;
      if (m < 49) {
        int r = rowmap[m];
        outh[r * 384 + h * 32 + j * 16 + cn] = (_Float16)oAcc[j][v];
      }
    }
}

// ---------------------------------------------------------------- launch
extern "C" void kernel_launch(void* const* d_in, const int* in_sizes, int n_in,
                              void* d_out, int out_size, void* d_ws, size_t ws_size,
                              hipStream_t stream) {
  (void)in_sizes; (void)n_in; (void)out_size; (void)ws_size;
  const float* x      = (const float*)d_in[0];   // [32, 3136, 384]
  const float* qkv_w  = (const float*)d_in[1];   // [1152, 384]
  const float* qkv_b  = (const float*)d_in[2];   // [1152]
  const float* proj_w = (const float*)d_in[3];   // [384, 384]
  const float* proj_b = (const float*)d_in[4];   // [384]
  const float* btab   = (const float*)d_in[5];   // [169, 12]

  const size_t M = 100352;  // 32 * 56 * 56

  _Float16* xh     = (_Float16*)d_ws;            // M*384 halves (reused for attn out)
  _Float16* wqkvh  = xh + M * 384;               // 1152*384
  _Float16* wprojh = wqkvh + (size_t)1152 * 384; // 384*384
  _Float16* qkvh   = wprojh + (size_t)384 * 384; // M*1152
  _Float16* attnh  = xh;                         // alias: x dead after QKV GEMM

  cast_f32_to_f16_v4<<<2048, 256, 0, stream>>>(x,      xh,     (int)(M * 384 / 4));
  cast_f32_to_f16_v4<<<128,  256, 0, stream>>>(qkv_w,  wqkvh,  1152 * 384 / 4);
  cast_f32_to_f16_v4<<<64,   256, 0, stream>>>(proj_w, wprojh, 384 * 384 / 4);

  gemm_qkv_f16<<<1568 * 18, 128, 0, stream>>>(xh, wqkvh, qkv_b, qkvh, 18);
  win_attn<<<2048 * 12, 128, 0, stream>>>(qkvh, btab, attnh);
  gemm_proj_f32<<<1568 * 6, 128, 0, stream>>>(attnh, wprojh, proj_b, (float*)d_out, 6);
}
